// HRRBindingSelfAttention_58634893525657
// MI455X (gfx1250) — compile-verified, matrix operand order checked by reading, not run on hardware
//
#include <hip/hip_runtime.h>
#include <hip/hip_bf16.h>
#include <cstdint>
#include <cstddef>

#define DMODEL 2048
#define LOG2D  11
#define NFREQ  1025
#define NB     4
#define SEQL   4096
#define NTOK   (NB * SEQL)      // 16384
#define NCHUNK 16
#define CSIZE  256              // SEQL / NCHUNK

typedef __attribute__((ext_vector_type(16))) __bf16 v16bf;
typedef __attribute__((ext_vector_type(8)))  float  v8f;

struct U16x16 { uint4 a; uint4 b; };   // 32 bytes == v16bf

// ---------------------------------------------------------------------------
// K0: convert w_out (fp32 [E][D]) -> bf16
// ---------------------------------------------------------------------------
__global__ void __launch_bounds__(256)
wconv_kernel(const float* __restrict__ w, __hip_bfloat16* __restrict__ wb) {
    int i = blockIdx.x * 256 + threadIdx.x;
    if (i < DMODEL * DMODEL) wb[i] = __float2bfloat16(w[i]);
}

// ---------------------------------------------------------------------------
// K1: rfft(2048) per token, radix-2 DIT in LDS (real input, imag=0).
// One workgroup (256 threads) per token.  Writes bins 0..1024 as float2.
// ---------------------------------------------------------------------------
__global__ void __launch_bounds__(256)
rfft_kernel(const float* __restrict__ x, float2* __restrict__ xhat) {
    __shared__ float re[DMODEL];
    __shared__ float im[DMODEL];
    const int t   = blockIdx.x;
    const int tid = threadIdx.x;
    const float* xt = x + (size_t)t * DMODEL;

    // bit-reversed load, imag = 0
    for (int i = tid; i < DMODEL; i += 256) {
        int j = (int)(__brev((unsigned)i) >> (32 - LOG2D));
        re[j] = xt[i];
        im[j] = 0.0f;
    }
    __syncthreads();

    for (int len = 2; len <= DMODEL; len <<= 1) {
        const int   half = len >> 1;
        const float angb = -6.28318530717958647692f / (float)len;
        for (int idx = tid; idx < (DMODEL / 2); idx += 256) {
            int blk = idx / half;
            int pos = idx - blk * half;
            int i0  = blk * len + pos;
            int i1  = i0 + half;
            float sn, cs;
            __sincosf(angb * (float)pos, &sn, &cs);
            float ur = re[i0], ui = im[i0];
            float vr = re[i1], vi = im[i1];
            float tr = vr * cs - vi * sn;
            float ti = vr * sn + vi * cs;
            re[i0] = ur + tr; im[i0] = ui + ti;
            re[i1] = ur - tr; im[i1] = ui - ti;
        }
        __syncthreads();
    }

    float2* o = xhat + (size_t)t * NFREQ;
    for (int f = tid; f < NFREQ; f += 256) o[f] = make_float2(re[f], im[f]);
}

// ---------------------------------------------------------------------------
// K2a: per-(b,f,chunk) partial sums of kv_f * xhat  (for the causal scan)
// ---------------------------------------------------------------------------
__global__ void __launch_bounds__(256)
scan_partial_kernel(const float2* __restrict__ xhat, const float* __restrict__ kvw,
                    float2* __restrict__ part) {
    int g = blockIdx.x * 256 + threadIdx.x;
    if (g >= NB * NFREQ * NCHUNK) return;
    int ch = g & (NCHUNK - 1);
    int bf = g >> 4;
    int f  = bf % NFREQ;
    int b  = bf / NFREQ;
    float kv = kvw[f];
    const float2* col = xhat + ((size_t)(b * SEQL + ch * CSIZE)) * NFREQ + f;
    float sr = 0.0f, si = 0.0f;
    for (int s = 0; s < CSIZE; ++s) {
        float2 v = col[(size_t)s * NFREQ];
        sr += v.x; si += v.y;
    }
    part[g] = make_float2(kv * sr, kv * si);
}

// ---------------------------------------------------------------------------
// K2b: exclusive chunk prefix + in-chunk inclusive scan; writes
//      qv = q_f * ( x_t  (complex*)  cumsum_{<=t}(kv_f * x) )   in place.
// ---------------------------------------------------------------------------
__global__ void __launch_bounds__(256)
scan_apply_kernel(float2* __restrict__ xhat, const float* __restrict__ qw,
                  const float* __restrict__ kvw, const float2* __restrict__ part) {
    int g = blockIdx.x * 256 + threadIdx.x;
    if (g >= NB * NFREQ * NCHUNK) return;
    int ch = g & (NCHUNK - 1);
    int bf = g >> 4;
    int f  = bf % NFREQ;
    int b  = bf / NFREQ;
    float q  = qw[f];
    float kv = kvw[f];

    float ar = 0.0f, ai = 0.0f;                 // exclusive prefix (already kv-weighted)
    int base = bf << 4;
    for (int c = 0; c < ch; ++c) {
        float2 p = part[base + c];
        ar += p.x; ai += p.y;
    }
    float2* col = xhat + ((size_t)(b * SEQL + ch * CSIZE)) * NFREQ + f;
    for (int s = 0; s < CSIZE; ++s) {
        float2 v = col[(size_t)s * NFREQ];
        ar += kv * v.x;                          // inclusive accumulate kv*x
        ai += kv * v.y;
        float zr = v.x * ar - v.y * ai;          // x_t (complex*) acc
        float zi = v.x * ai + v.y * ar;
        col[(size_t)s * NFREQ] = make_float2(q * zr, q * zi);
    }
}

// ---------------------------------------------------------------------------
// K3: irfft(2048) per token: Hermitian expand, bit-reverse permute, inverse
//     radix-2 stages, scale 1/N, store bf16 values_hat.
// ---------------------------------------------------------------------------
__global__ void __launch_bounds__(256)
irfft_kernel(const float2* __restrict__ qv, __hip_bfloat16* __restrict__ V) {
    __shared__ float re[DMODEL];
    __shared__ float im[DMODEL];
    const int t   = blockIdx.x;
    const int tid = threadIdx.x;
    const float2* q = qv + (size_t)t * NFREQ;

    for (int f = tid; f < NFREQ; f += 256) {
        float2 v = q[f];
        re[f] = v.x; im[f] = v.y;
        if (f > 0 && f < DMODEL / 2) {           // Hermitian symmetry
            re[DMODEL - f] =  v.x;
            im[DMODEL - f] = -v.y;
        }
    }
    __syncthreads();

    // in-place bit-reverse permutation
    for (int i = tid; i < DMODEL; i += 256) {
        int j = (int)(__brev((unsigned)i) >> (32 - LOG2D));
        if (j > i) {
            float tr = re[i], ti = im[i];
            re[i] = re[j]; im[i] = im[j];
            re[j] = tr;    im[j] = ti;
        }
    }
    __syncthreads();

    for (int len = 2; len <= DMODEL; len <<= 1) {
        const int   half = len >> 1;
        const float angb = 6.28318530717958647692f / (float)len;   // inverse sign
        for (int idx = tid; idx < (DMODEL / 2); idx += 256) {
            int blk = idx / half;
            int pos = idx - blk * half;
            int i0  = blk * len + pos;
            int i1  = i0 + half;
            float sn, cs;
            __sincosf(angb * (float)pos, &sn, &cs);
            float ur = re[i0], ui = im[i0];
            float vr = re[i1], vi = im[i1];
            float tr = vr * cs - vi * sn;
            float ti = vr * sn + vi * cs;
            re[i0] = ur + tr; im[i0] = ui + ti;
            re[i1] = ur - tr; im[i1] = ui - ti;
        }
        __syncthreads();
    }

    __hip_bfloat16* o = V + (size_t)t * DMODEL;
    const float scale = 1.0f / (float)DMODEL;
    for (int d = tid; d < DMODEL; d += 256) o[d] = __float2bfloat16(re[d] * scale);
}

// ---------------------------------------------------------------------------
// K4: out[t][e] = sum_d V[t][d] * Wb[e][d]   (bf16 WMMA, fp32 accumulate)
// 256 threads = 8 waves; BM=BN=128, BK=32; waves 4(M) x 2(N), each wave owns
// 2x4 16x16 accumulator tiles.  Global->LDS staging is double-buffered with
// CDNA5 async copies (global_load_async_to_lds_b128 / s_wait_asynccnt).
// ---------------------------------------------------------------------------
__device__ __forceinline__ v16bf load_frag(const unsigned short* base, int row, int hi8) {
    const unsigned short* p = base + row * 32 + hi8;   // 16B aligned (hi8 in {0,8})
    U16x16 u;
    u.a = *(const uint4*)(p);        // K = hi8 .. hi8+7
    u.b = *(const uint4*)(p + 16);   // K = hi8+16 .. hi8+23
    return __builtin_bit_cast(v16bf, u);
}

__global__ void __launch_bounds__(256)
gemm_kernel(const unsigned short* __restrict__ V, const unsigned short* __restrict__ Wb,
            float* __restrict__ out) {
    // LDS: two stages of (A:128x32 + B:128x32) bf16 tiles = 32 KB
    __shared__ __align__(16) unsigned short smem[2 * 8192];

    const int tid  = threadIdx.x;
    const int lane = tid & 31;
    const int wave = tid >> 5;
    const int lr   = lane & 15;
    const int hi   = lane >> 4;           // lane group (K-split per ISA layout)
    const int hi8  = hi * 8;
    const int wm   = wave & 3;            // wave M position (0..3) -> 32 rows
    const int wn   = wave >> 2;           // wave N position (0..1) -> 64 cols
    const int row0 = (int)blockIdx.x * 128;
    const int col0 = (int)blockIdx.y * 128;

    const uint32_t ldsBase = (uint32_t)(uintptr_t)(&smem[0]);

    v8f acc[2][4];
#pragma unroll
    for (int mt = 0; mt < 2; ++mt)
#pragma unroll
        for (int nt = 0; nt < 4; ++nt) acc[mt][nt] = {};

    auto issue_stage = [&](int kt, int buf) {
        const size_t   kb = (size_t)kt * 32;
        const uint32_t bb = ldsBase + (uint32_t)buf * 16384u;
#pragma unroll
        for (int i = 0; i < 2; ++i) {
            int tr = tid + i * 256;       // 0..511
            int rr = tr >> 2;             // tile row 0..127
            int cc = tr & 3;              // 16B segment 0..3
            // A tile: values_hat[row0+rr][kb + cc*8 ..]
            uint64_t ga = (uint64_t)(uintptr_t)(V + (size_t)(row0 + rr) * DMODEL + kb + (size_t)cc * 8);
            uint32_t la = bb + (uint32_t)(rr * 64 + cc * 16);
            asm volatile("global_load_async_to_lds_b128 %0, %1, off"
                         :: "v"(la), "v"(ga) : "memory");
            // B tile: Wb[col0+rr][kb + cc*8 ..]  (column n of B contiguous in K)
            uint64_t gb = (uint64_t)(uintptr_t)(Wb + (size_t)(col0 + rr) * DMODEL + kb + (size_t)cc * 8);
            uint32_t lb = bb + 8192u + (uint32_t)(rr * 64 + cc * 16);
            asm volatile("global_load_async_to_lds_b128 %0, %1, off"
                         :: "v"(lb), "v"(gb) : "memory");
        }
    };

    const int nIter = DMODEL / 32;        // 64
    issue_stage(0, 0);

    for (int kt = 0; kt < nIter; ++kt) {
        if (kt + 1 < nIter) {
            issue_stage(kt + 1, (kt + 1) & 1);
            asm volatile("s_wait_asynccnt 4" ::: "memory");  // stage kt complete
        } else {
            asm volatile("s_wait_asynccnt 0" ::: "memory");
        }
        __syncthreads();

        const unsigned short* As = smem + (kt & 1) * 8192;
        const unsigned short* Bs = As + 4096;

        v16bf a[2], b[4];
#pragma unroll
        for (int mt = 0; mt < 2; ++mt)
            a[mt] = load_frag(As, wm * 32 + mt * 16 + lr, hi8);
#pragma unroll
        for (int nt = 0; nt < 4; ++nt)
            b[nt] = load_frag(Bs, wn * 64 + nt * 16 + lr, hi8);

#pragma unroll
        for (int mt = 0; mt < 2; ++mt)
#pragma unroll
            for (int nt = 0; nt < 4; ++nt)
                acc[mt][nt] = __builtin_amdgcn_wmma_f32_16x16x32_bf16(
                    /*neg_a=*/false, a[mt], /*neg_b=*/false, b[nt],
                    /*c_mod=*/(short)0, acc[mt][nt],
                    /*reuse_a=*/false, /*reuse_b=*/false);

        __syncthreads();
    }

    // C/D layout: VGPR r -> (M = r + hi*8, N = lr) within each 16x16 tile
#pragma unroll
    for (int mt = 0; mt < 2; ++mt)
#pragma unroll
        for (int nt = 0; nt < 4; ++nt)
#pragma unroll
            for (int rr = 0; rr < 8; ++rr) {
                int grow = row0 + wm * 32 + mt * 16 + hi * 8 + rr;
                int gcol = col0 + wn * 64 + nt * 16 + lr;
                out[(size_t)grow * DMODEL + gcol] = acc[mt][nt][rr];
            }
}

// ---------------------------------------------------------------------------
// Host-side launch
// ---------------------------------------------------------------------------
extern "C" void kernel_launch(void* const* d_in, const int* in_sizes, int n_in,
                              void* d_out, int out_size, void* d_ws, size_t ws_size,
                              hipStream_t stream) {
    (void)in_sizes; (void)n_in; (void)out_size; (void)ws_size;

    const float* x   = (const float*)d_in[0];   // [4,4096,2048]
    const float* qw  = (const float*)d_in[1];   // [1,1,1025]
    const float* kvw = (const float*)d_in[2];   // [1,1,1025]
    const float* w   = (const float*)d_in[3];   // [2048,2048]
    float*       out = (float*)d_out;           // [4,4096,2048]

    // workspace layout (bytes, all 256-aligned)
    char* ws = (char*)d_ws;
    const size_t xhat_bytes = (size_t)NTOK * NFREQ * sizeof(float2);     // 134,348,800
    const size_t part_bytes = (size_t)NB * NFREQ * NCHUNK * sizeof(float2); // 524,800
    const size_t wb_bytes   = (size_t)DMODEL * DMODEL * sizeof(unsigned short); // 8 MiB
    float2*          xhat = (float2*)(ws);
    float2*          part = (float2*)(ws + xhat_bytes);
    __hip_bfloat16*  wb   = (__hip_bfloat16*)(ws + xhat_bytes + part_bytes);
    __hip_bfloat16*  vb   = (__hip_bfloat16*)(ws + xhat_bytes + part_bytes + wb_bytes);

    // K0: W -> bf16
    wconv_kernel<<<(DMODEL * DMODEL) / 256, 256, 0, stream>>>(w, wb);

    // K1: rfft per token
    rfft_kernel<<<NTOK, 256, 0, stream>>>(x, xhat);

    // K2: chunked causal scan in frequency domain
    const int nscan = NB * NFREQ * NCHUNK;              // 65,600
    const int sblk  = (nscan + 255) / 256;
    scan_partial_kernel<<<sblk, 256, 0, stream>>>(xhat, kvw, part);
    scan_apply_kernel<<<sblk, 256, 0, stream>>>(xhat, qw, kvw, part);

    // K3: irfft per token -> bf16 values_hat
    irfft_kernel<<<NTOK, 256, 0, stream>>>(xhat, vb);

    // K4: WMMA GEMM  out = values_hat @ W^T
    dim3 grid(NTOK / 128, DMODEL / 128);                // 128 x 16
    gemm_kernel<<<grid, 256, 0, stream>>>((const unsigned short*)vb,
                                          (const unsigned short*)wb, out);
}
